// NeuralMemoryBank_12128987643985
// MI455X (gfx1250) — matrix operand order repaired and checked by reading, not compiled, not run on hardware
//
#include <hip/hip_runtime.h>
#include <hip/hip_bf16.h>
#include <stdint.h>

// ---------------------------------------------------------------------------
// NeuralMemoryBank fused kernel for gfx1250 (MI455X), wave32 + WMMA bf16.
//   B=4, S=2048, H=1024, M=8192
// Pipeline:
//   K0: f32 -> bf16 converts (x, memory, W) + memory transpose (memT)
//   K1: proj GEMM (out = A @ W^T + b) in bf16 via v_wmma_f32_16x16x32_bf16
//   K2: fused scores/softmax/read (flash-style, 2-pass softmax, deterministic)
//       - Q tile staged to LDS with global_load_async_to_lds_b128 (ASYNCcnt)
//       - PK stream prefetched with global_prefetch_b8
//   K3: mean over S (deterministic fixed-order reduction)
// ---------------------------------------------------------------------------

#define HDIM 1024
#define MDIM 8192
#define RDIM 8192   // B*S
#define SSEQ 2048
#define BATCH 4

typedef __attribute__((ext_vector_type(16))) __bf16 v16bf;
typedef __attribute__((ext_vector_type(8)))  float  v8f;

union Frag {
    v16bf v;
    unsigned int u[8];
};

__device__ __forceinline__ unsigned short f2bf(float f) {
    unsigned int u = __float_as_uint(f);
    unsigned int r = u + 0x7FFFu + ((u >> 16) & 1u);   // round-to-nearest-even
    return (unsigned short)(r >> 16);
}

__device__ __forceinline__ v8f wmma_bf16(const Frag& a, const Frag& b, v8f c) {
    return __builtin_amdgcn_wmma_f32_16x16x32_bf16(
        false, a.v, false, b.v, (short)0, c, false, false);
}

// A-fragment (16x32 bf16) K index for VGPR v, lane-group lg (doc 7.12.2)
__device__ __forceinline__ int a_k(int v, int lg) {
    return ((v < 4) ? 2 * v : 16 + 2 * (v - 4)) + lg * 8;
}
// B-fragment (32x16 bf16) K index for VGPR v, lane-group lg (same order as A)
__device__ __forceinline__ int b_k(int v, int lg) {
    return ((v < 4) ? 2 * v : 16 + 2 * (v - 4)) + lg * 8;
}

// Async 16B copy: global -> LDS, tracked on ASYNCcnt (gfx1250).
__device__ __forceinline__ void async_g2l_b128(unsigned int lds_byte_off,
                                               unsigned long long gaddr) {
    asm volatile("global_load_async_to_lds_b128 %0, %1, off"
                 :: "v"(lds_byte_off), "v"(gaddr)
                 : "memory");
}
__device__ __forceinline__ void wait_asynccnt0() {
    asm volatile("s_wait_asynccnt 0x0" ::: "memory");
}

// --------------------------- K0: conversions -------------------------------

__global__ void cvt_bf16_kernel(const float* __restrict__ src,
                                unsigned short* __restrict__ dst, int n) {
    int i = blockIdx.x * blockDim.x + threadIdx.x;
    if (i < n) dst[i] = f2bf(src[i]);
}

__global__ void transpose_bf16_kernel(const float* __restrict__ mem,
                                      unsigned short* __restrict__ memT) {
    // mem [MDIM][HDIM] f32 -> memT [HDIM][MDIM] bf16
    int m = blockIdx.x * blockDim.x + threadIdx.x;
    int h = blockIdx.y;
    memT[(size_t)h * MDIM + m] = f2bf(mem[(size_t)m * HDIM + h]);
}

// --------------------------- K1: projection GEMM ---------------------------
// out[r][h] = sum_k A[r][k] * W[h][k] + bias[h]   (bf16 out, f32 accum)
__global__ __launch_bounds__(256)
void proj_gemm_kernel(const unsigned short* __restrict__ A16,  // [rows][H] bf16
                      const unsigned short* __restrict__ W16,  // [H][H] bf16
                      const float* __restrict__ bias,          // [H]
                      unsigned short* __restrict__ out,        // [rows][H] bf16
                      int rows) {
    const int colTiles = HDIM / 16;  // 64
    int gwave = (blockIdx.x * blockDim.x + threadIdx.x) >> 5;
    int lane = threadIdx.x & 31;
    int lg = lane >> 4, ln = lane & 15;
    int r0 = (gwave / colTiles) * 16;
    int h0 = (gwave % colTiles) * 16;
    if (r0 >= rows) return;

    v8f acc = {};
    for (int k0 = 0; k0 < HDIM; k0 += 32) {
        Frag a, b;
#pragma unroll
        for (int v = 0; v < 8; ++v) {
            a.u[v] = *(const unsigned int*)(A16 + (size_t)(r0 + ln) * HDIM + k0 + a_k(v, lg));
            b.u[v] = *(const unsigned int*)(W16 + (size_t)(h0 + ln) * HDIM + k0 + b_k(v, lg));
        }
        acc = wmma_bf16(a, b, acc);
    }
    float bv = bias[h0 + ln];
#pragma unroll
    for (int v = 0; v < 8; ++v) {
        out[(size_t)(r0 + v + 8 * lg) * HDIM + h0 + ln] = f2bf(acc[v] + bv);
    }
}

// --------------------------- K2: fused attention ---------------------------
// One 128-thread workgroup (4 waves) per 16-row query block.
//  Pass A: softmax stats (max, sum) over M, waves split M, merged in LDS.
//  Pass B: per 512-wide m-chunk: waves compute normalized P -> LDS, then each
//          wave accumulates its 256-column H slice of read = P @ memory.
__global__ __launch_bounds__(128)
void attn_fused_kernel(const unsigned short* __restrict__ Q,    // [RDIM][H] bf16
                       const unsigned short* __restrict__ PK,   // [MDIM][H] bf16
                       const unsigned short* __restrict__ MT,   // [H][MDIM] bf16
                       float* __restrict__ Rd)                  // [RDIM][H] f32
{
    __shared__ unsigned short q_lds[16 * HDIM];   // 32 KB
    __shared__ unsigned short p_lds[16 * 512];    // 16 KB
    __shared__ float stats[4][16][2];
    __shared__ float merged[16][2];               // {max, 1/den}

    const int tid = threadIdx.x;
    const int wave = tid >> 5, lane = tid & 31;
    const int lg = lane >> 4, ln = lane & 15;
    const int r0 = blockIdx.x * 16;
    const float scale = 0.03125f;  // 1/sqrt(H) = 1/32

    // stage 16x1024 bf16 Q tile into LDS via async DMA (32 KB = 2048 x b128)
    {
        const unsigned int q_lds_base = (unsigned int)(size_t)q_lds;
        const unsigned long long gbase =
            (unsigned long long)(const void*)(Q + (size_t)r0 * HDIM);
        for (int i = tid; i < (16 * HDIM * 2) / 16; i += 128) {
            async_g2l_b128(q_lds_base + (unsigned int)(i * 16),
                           gbase + (unsigned long long)(i * 16));
        }
        wait_asynccnt0();
    }
    __syncthreads();

    // -------- Pass A: softmax stats, wave handles m in [wave*2048, +2048)
    float rmax[8], rsum[8];
#pragma unroll
    for (int v = 0; v < 8; ++v) { rmax[v] = -3.0e38f; rsum[v] = 0.0f; }

    for (int m0 = wave * 2048; m0 < (wave + 1) * 2048; m0 += 16) {
        // pull next key tile toward the WGP while this tile's WMMAs run
        if (m0 + 16 < MDIM)
            __builtin_prefetch(PK + (size_t)(m0 + 16 + ln) * HDIM, 0, 0);
        v8f d = {};
        for (int k0 = 0; k0 < HDIM; k0 += 32) {
            Frag a, b;
#pragma unroll
            for (int v = 0; v < 8; ++v) {
                a.u[v] = *(const unsigned int*)(q_lds + ln * HDIM + k0 + a_k(v, lg));
                b.u[v] = *(const unsigned int*)(PK + (size_t)(m0 + ln) * HDIM + k0 + b_k(v, lg));
            }
            d = wmma_bf16(a, b, d);
        }
#pragma unroll
        for (int v = 0; v < 8; ++v) {   // row = v + 8*lg; reduce over 16-lane group
            float s = d[v] * scale;
            float t = s;
            t = fmaxf(t, __shfl_xor(t, 1, 32));
            t = fmaxf(t, __shfl_xor(t, 2, 32));
            t = fmaxf(t, __shfl_xor(t, 4, 32));
            t = fmaxf(t, __shfl_xor(t, 8, 32));
            float nm = fmaxf(rmax[v], t);
            float e = __expf(s - nm);
            e += __shfl_xor(e, 1, 32);
            e += __shfl_xor(e, 2, 32);
            e += __shfl_xor(e, 4, 32);
            e += __shfl_xor(e, 8, 32);
            rsum[v] = rsum[v] * __expf(rmax[v] - nm) + e;
            rmax[v] = nm;
        }
    }
    if (ln == 0) {
#pragma unroll
        for (int v = 0; v < 8; ++v) {
            stats[wave][v + 8 * lg][0] = rmax[v];
            stats[wave][v + 8 * lg][1] = rsum[v];
        }
    }
    __syncthreads();
    if (tid < 16) {   // deterministic fixed-order merge over the 4 waves
        float mm = stats[0][tid][0];
        for (int w = 1; w < 4; ++w) mm = fmaxf(mm, stats[w][tid][0]);
        float den = 0.0f;
        for (int w = 0; w < 4; ++w) den += stats[w][tid][1] * __expf(stats[w][tid][0] - mm);
        merged[tid][0] = mm;
        merged[tid][1] = 1.0f / den;
    }
    __syncthreads();

    // -------- Pass B: O accumulation; this wave owns H columns [wave*256, +256)
    v8f o[16];
    v8f zero = {};
#pragma unroll
    for (int t = 0; t < 16; ++t) o[t] = zero;
    const int hbase = wave * 256;

    for (int c0 = 0; c0 < MDIM; c0 += 512) {
        // B1: this wave produces P for m in [c0 + wave*128, +128)
        for (int mt = 0; mt < 8; ++mt) {
            int m0 = c0 + wave * 128 + mt * 16;
            if (m0 + 16 < MDIM)
                __builtin_prefetch(PK + (size_t)(m0 + 16 + ln) * HDIM, 0, 0);
            v8f d = {};
            for (int k0 = 0; k0 < HDIM; k0 += 32) {
                Frag a, b;
#pragma unroll
                for (int v = 0; v < 8; ++v) {
                    a.u[v] = *(const unsigned int*)(q_lds + ln * HDIM + k0 + a_k(v, lg));
                    b.u[v] = *(const unsigned int*)(PK + (size_t)(m0 + ln) * HDIM + k0 + b_k(v, lg));
                }
                d = wmma_bf16(a, b, d);
            }
#pragma unroll
            for (int v = 0; v < 8; ++v) {
                int row = v + 8 * lg;
                float p = __expf(d[v] * scale - merged[row][0]) * merged[row][1];
                p_lds[row * 512 + (wave * 128 + mt * 16 + ln)] = f2bf(p);
            }
        }
        __syncthreads();
        // B2: O[t] += P(16x32) @ memT(32x16) over the chunk, 16 H-tiles resident
        for (int ks = 0; ks < 512; ks += 32) {
            Frag a;
#pragma unroll
            for (int v = 0; v < 8; ++v) {
                a.u[v] = *(const unsigned int*)(p_lds + ln * 512 + ks + a_k(v, lg));
            }
#pragma unroll
            for (int t = 0; t < 16; ++t) {
                Frag b;
#pragma unroll
                for (int v = 0; v < 8; ++v) {
                    b.u[v] = *(const unsigned int*)(MT + (size_t)(hbase + t * 16 + ln) * MDIM + c0 + ks + b_k(v, lg));
                }
                o[t] = wmma_bf16(a, b, o[t]);
            }
        }
        __syncthreads();   // protect p_lds before next chunk overwrites it
    }

    // store read rows (f32)
#pragma unroll
    for (int t = 0; t < 16; ++t) {
#pragma unroll
        for (int v = 0; v < 8; ++v) {
            Rd[(size_t)(r0 + v + 8 * lg) * HDIM + hbase + t * 16 + ln] = o[t][v];
        }
    }
}

// --------------------------- K3: mean over S -------------------------------

__global__ void reduce_mean_kernel(const float* __restrict__ Rd,
                                   float* __restrict__ out) {
    int h = blockIdx.x * blockDim.x + threadIdx.x;  // 0..1023
    int b = blockIdx.y;                             // 0..3
    if (h >= HDIM) return;
    const float* p = Rd + (size_t)b * SSEQ * HDIM + h;
    float acc = 0.0f;
    for (int s = 0; s < SSEQ; ++s) acc += p[(size_t)s * HDIM];
    out[b * HDIM + h] = acc * (1.0f / (float)SSEQ);
}

// --------------------------- launch ----------------------------------------

extern "C" void kernel_launch(void* const* d_in, const int* in_sizes, int n_in,
                              void* d_out, int out_size, void* d_ws, size_t ws_size,
                              hipStream_t stream) {
    const float* x    = (const float*)d_in[0];  // [4,2048,1024]
    const float* mem  = (const float*)d_in[1];  // [8192,1024]
    const float* W    = (const float*)d_in[2];  // [1024,1024]
    const float* bias = (const float*)d_in[3];  // [1024]
    float* out = (float*)d_out;                 // [4,1024]
    (void)in_sizes; (void)n_in; (void)out_size; (void)ws_size;

    char* ws = (char*)d_ws;
    unsigned short* x16   = (unsigned short*)(ws + ((size_t)0  << 20));  // 16 MB
    unsigned short* mem16 = (unsigned short*)(ws + ((size_t)16 << 20));  // 16 MB
    unsigned short* W16   = (unsigned short*)(ws + ((size_t)32 << 20));  //  2 MB
    unsigned short* memT  = (unsigned short*)(ws + ((size_t)34 << 20));  // 16 MB
    unsigned short* q16   = (unsigned short*)(ws + ((size_t)50 << 20));  // 16 MB
    unsigned short* pk16  = (unsigned short*)(ws + ((size_t)66 << 20));  // 16 MB
    float*          rd    = (float*)         (ws + ((size_t)82 << 20));  // 32 MB

    // K0: conversions
    cvt_bf16_kernel<<<(RDIM * HDIM) / 256, 256, 0, stream>>>(x, x16, RDIM * HDIM);
    cvt_bf16_kernel<<<(MDIM * HDIM) / 256, 256, 0, stream>>>(mem, mem16, MDIM * HDIM);
    cvt_bf16_kernel<<<(HDIM * HDIM) / 256, 256, 0, stream>>>(W, W16, HDIM * HDIM);
    transpose_bf16_kernel<<<dim3(MDIM / 256, HDIM), 256, 0, stream>>>(mem, memT);

    // K1: projections (queries and projected keys), one 16x16 tile per wave
    {
        int waves = (RDIM / 16) * (HDIM / 16);          // 32768
        int blocks = (waves * 32) / 256;                // 4096
        proj_gemm_kernel<<<blocks, 256, 0, stream>>>(x16,   W16, bias, q16,  RDIM);
        proj_gemm_kernel<<<blocks, 256, 0, stream>>>(mem16, W16, bias, pk16, MDIM);
    }

    // K2: fused scores/softmax/read
    attn_fused_kernel<<<RDIM / 16, 128, 0, stream>>>(q16, pk16, memT, rd);

    // K3: deterministic mean over S
    reduce_mean_kernel<<<dim3(HDIM / 256, BATCH), 256, 0, stream>>>(rd, out);
}
